// ONLSTMStack_76725295776452
// MI455X (gfx1250) — matrix operand order repaired and checked by reading, not compiled, not run on hardware
//
#include <hip/hip_runtime.h>
#include <hip/hip_bf16.h>

// ---------------- problem constants ----------------
#define Bq    32
#define Tq    512
#define INq   512
#define Hq    1024
#define NCHq  16
#define CHUNKq 64
#define GDIMq 4128           // 4*H + 2*NCH
#define NTILES_G (GDIMq/16)  // 258 N-tiles of 16
#define MROWS   (Bq*Tq)      // 16384
#define NPAIRS  (NTILES_G/2) // 129 N-pairs (32 cols each)
#define MBLKS   (MROWS/64)   // 256 M-blocks (64 rows each)
#define NGRP2   ((NPAIRS+7)/8) // 17 wave-groups of 8 N-pairs
#define KSPLIT  4            // deterministic K-split of the recurrent GEMM
#define KCHUNK  (Hq/KSPLIT)  // 256

typedef __bf16 v16bf __attribute__((ext_vector_type(16)));
typedef __bf16 v8bf  __attribute__((ext_vector_type(8)));
typedef float  v8f   __attribute__((ext_vector_type(8)));

__device__ __forceinline__ float sigf(float x) { return 1.0f / (1.0f + __expf(-x)); }

// Build a 16-element bf16 A/B fragment from two contiguous 16B chunks (ISA lane layout).
__device__ __forceinline__ v16bf load_frag(const __bf16* p) {
    v8bf a0 = *(const v8bf*)(p);
    v8bf a1 = *(const v8bf*)(p + 16);
    v16bf a;
#pragma unroll
    for (int i = 0; i < 8; ++i) { a[i] = a0[i]; a[i + 8] = a1[i]; }
    return a;
}

__device__ __forceinline__ v8f wmma_bf16(v16bf a, v16bf b, v8f c) {
    return __builtin_amdgcn_wmma_f32_16x16x32_bf16(false, a, false, b, (short)0, c, false, false);
}

// ---------------- init / convert ----------------
__global__ void zero_init_kernel(float* c0, float* c1, __bf16* ybf0, __bf16* ybf1) {
    int idx = blockIdx.x * blockDim.x + threadIdx.x;          // 0 .. B*H-1
    if (idx >= Bq * Hq) return;
    int b = idx >> 10, j = idx & 1023;
    c0[idx] = 0.0f;
    c1[idx] = 0.0f;
    size_t o = ((size_t)b * (Tq + 1)) * Hq + j;               // t=0 column
    ybf0[o] = (__bf16)0.0f;
    ybf1[o] = (__bf16)0.0f;
}

__global__ void f32_to_bf16_kernel(const float* __restrict__ src, __bf16* __restrict__ dst, int n) {
    int idx = blockIdx.x * blockDim.x + threadIdx.x;
    if (idx < n) dst[idx] = (__bf16)src[idx];
}

// ---------------- input-projection GEMM: t_in = X * W_ih^T + b_ih ----------------
// Register-blocked 4M x 2N per wave: 64x32 output tile, 8 WMMAs per K=32 step.
// (plain loop body: LLVM's clause scheduler handles load/WMMA overlap without spills)
__global__ void gemm_tin_kernel(const __bf16* __restrict__ A,
                                const __bf16* __restrict__ W,
                                const float*  __restrict__ bias,
                                float* __restrict__ out,
                                int K, int skip) {
    const int lane  = threadIdx.x & 31;
    const int wave  = threadIdx.x >> 5;
    const int mBlk  = blockIdx.x / NGRP2;
    const int grp   = blockIdx.x % NGRP2;
    const int nPair = grp * 8 + wave;
    if (nPair >= NPAIRS) return;

    const int half = lane >> 4;                  // 0: lanes 0-15, 1: lanes 16-31
    const int lrow = lane & 15;

    const __bf16* ap[4];
#pragma unroll
    for (int mi = 0; mi < 4; ++mi) {
        int m = mBlk * 64 + mi * 16 + lrow;
        ap[mi] = A + ((size_t)(m / Tq) * (Tq + skip) + (m % Tq) + skip) * (size_t)K + half * 8;
    }
    const __bf16* wp0 = W + (size_t)((nPair * 2 + 0) * 16 + lrow) * K + half * 8;
    const __bf16* wp1 = W + (size_t)((nPair * 2 + 1) * 16 + lrow) * K + half * 8;

    v8f acc[4][2];
#pragma unroll
    for (int mi = 0; mi < 4; ++mi) { acc[mi][0] = (v8f){}; acc[mi][1] = (v8f){}; }

    for (int kb = 0; kb < K; kb += 32) {
        v16bf b0 = load_frag(wp0 + kb);
        v16bf b1 = load_frag(wp1 + kb);
#pragma unroll
        for (int mi = 0; mi < 4; ++mi) {
            v16bf a = load_frag(ap[mi] + kb);
            acc[mi][0] = wmma_bf16(a, b0, acc[mi][0]);
            acc[mi][1] = wmma_bf16(a, b1, acc[mi][1]);
        }
    }

#pragma unroll
    for (int ni = 0; ni < 2; ++ni) {
        const int g  = (nPair * 2 + ni) * 16 + lrow;
        const float bg = bias[g];
#pragma unroll
        for (int mi = 0; mi < 4; ++mi) {
#pragma unroll
            for (int r = 0; r < 8; ++r) {
                int mr = mBlk * 64 + mi * 16 + r + half * 8;
                out[(size_t)mr * GDIMq + g] = acc[mi][ni][r] + bg;
            }
        }
    }
}

// ---------------- recurrent step GEMM (K-split partials) ----------------
// gridDim.y = KSPLIT deterministic K-chunks; each wave: 2M x 1N tile over K=256.
// Writes partial sums to gpart (KSPLIT, B, GDIM); gate kernel reduces in fixed order.
__global__ void step_gemm_kernel(const __bf16* __restrict__ ybf,
                                 const __bf16* __restrict__ Whh,
                                 const float*  __restrict__ tin,
                                 float* __restrict__ gpart,
                                 int t) {
    const int lane  = threadIdx.x & 31;
    const int nTile = blockIdx.x * 8 + (threadIdx.x >> 5);
    if (nTile >= NTILES_G) return;               // 258 N-tiles
    const int kc    = blockIdx.y;                // K chunk
    const int kOff  = kc * KCHUNK;

    const int half = lane >> 4;
    const int lrow = lane & 15;

    // Prefetch next step's t_in slice (HBM-resident) once per N-tile.
    if (kc == 0 && t + 1 < Tq) {
        int prow = lrow + 16 * half;             // 32 lanes cover all 32 batch rows
        const float* pf = tin + ((size_t)prow * Tq + (t + 1)) * GDIMq + nTile * 16;
        __builtin_prefetch(pf, 0, 1);
    }

    const __bf16* hp0 = ybf + ((size_t)(lrow)      * (Tq + 1) + t) * Hq + kOff + half * 8;
    const __bf16* hp1 = ybf + ((size_t)(16 + lrow) * (Tq + 1) + t) * Hq + kOff + half * 8;
    const __bf16* wp  = Whh + (size_t)(nTile * 16 + lrow) * Hq + kOff + half * 8;

    v8f acc0 = {}, acc1 = {};
    for (int kb = 0; kb < KCHUNK; kb += 32) {
        v16bf bm = load_frag(wp + kb);
        v16bf a0 = load_frag(hp0 + kb);
        v16bf a1 = load_frag(hp1 + kb);
        acc0 = wmma_bf16(a0, bm, acc0);
        acc1 = wmma_bf16(a1, bm, acc1);
    }

    const int g = nTile * 16 + lrow;
    float* gp = gpart + (size_t)kc * Bq * GDIMq;
#pragma unroll
    for (int r = 0; r < 8; ++r) {
        int b0 = r + half * 8;
        gp[(size_t)b0        * GDIMq + g] = acc0[r];
        gp[(size_t)(16 + b0) * GDIMq + g] = acc1[r];
    }
}

// ---------------- gate nonlinearity + state update ----------------
// Reduces the K-split partials (fixed order) + t_in + b_hh, then applies ON-LSTM cell.
__device__ __forceinline__ float gate_val(const float* gpart, const float* tin_row,
                                          const float* bhh, int b, int g) {
    float v = tin_row[g] + bhh[g];
#pragma unroll
    for (int kc = 0; kc < KSPLIT; ++kc)
        v += gpart[((size_t)kc * Bq + b) * GDIMq + g];
    return v;
}

__global__ void gate_step_kernel(const float* __restrict__ gpart,
                                 const float* __restrict__ tin,
                                 const float* __restrict__ bhh,
                                 float* __restrict__ c,        // (B, H) in-place
                                 float* __restrict__ hf,       // (B, H) f32 h state
                                 __bf16* __restrict__ ybf,     // (B, T+1, H) bf16 recurrent
                                 float* __restrict__ out_seq,  // (B, T, H) or nullptr
                                 int t) {
    const int b   = blockIdx.x;
    const int tid = threadIdx.x;                 // 0 .. 1023
    const float* tin_row = tin + ((size_t)b * Tq + t) * GDIMq;
    __shared__ float sci[NCHq];                  // cingate = 1 - cumsum(softmax(head0))
    __shared__ float scf[NCHq];                  // cforget =     cumsum(softmax(head1))

    if (tid < 2) {
        float gv[NCHq];
        float mx = -3.4e38f;
#pragma unroll
        for (int i = 0; i < NCHq; ++i) {
            gv[i] = gate_val(gpart, tin_row, bhh, b, tid * NCHq + i);
            mx = fmaxf(mx, gv[i]);
        }
        float e[NCHq], s = 0.0f;
#pragma unroll
        for (int i = 0; i < NCHq; ++i) { e[i] = __expf(gv[i] - mx); s += e[i]; }
        float inv = 1.0f / s, cum = 0.0f;
#pragma unroll
        for (int i = 0; i < NCHq; ++i) {
            cum += e[i] * inv;
            if (tid == 0) sci[i] = 1.0f - cum;
            else          scf[i] = cum;
        }
    }
    __syncthreads();

    const int n = tid >> 6;                      // chunk index
    const int j = tid & 63;                      // element in chunk
    const int gb = 2 * NCHq;                     // start of chunked gates
    float og   = sigf(gate_val(gpart, tin_row, bhh, b, gb + (0 * NCHq + n) * CHUNKq + j));
    float cell = tanhf(gate_val(gpart, tin_row, bhh, b, gb + (1 * NCHq + n) * CHUNKq + j));
    float ig   = sigf(gate_val(gpart, tin_row, bhh, b, gb + (2 * NCHq + n) * CHUNKq + j));
    float fg   = sigf(gate_val(gpart, tin_row, bhh, b, gb + (3 * NCHq + n) * CHUNKq + j));

    float ci = sci[n], cf = scf[n];
    float ov = cf * ci;
    fg = fg * ov + (cf - ov);
    ig = ig * ov + (ci - ov);

    const size_t idx = (size_t)b * Hq + tid;
    float cy = fg * c[idx] + ig * cell;
    c[idx] = cy;
    float hy = og * tanhf(cy);
    hf[idx] = hy;
    ybf[((size_t)b * (Tq + 1) + t + 1) * Hq + tid] = (__bf16)hy;
    if (out_seq) out_seq[((size_t)b * Tq + t) * Hq + tid] = hy;
}

// ---------------- final state copy-out ----------------
__global__ void copy_out_kernel(const float* hf0, const float* hf1,
                                const float* c0, const float* c1,
                                float* dst) {
    const int NHS = Bq * Hq;                     // 32768
    int idx = blockIdx.x * blockDim.x + threadIdx.x;
    if (idx >= 4 * NHS) return;
    if      (idx <     NHS) dst[idx] = hf0[idx];
    else if (idx < 2 * NHS) dst[idx] = hf1[idx - NHS];
    else if (idx < 3 * NHS) dst[idx] = c0[idx - 2 * NHS];
    else                    dst[idx] = c1[idx - 3 * NHS];
}

// ---------------- host side ----------------
static inline size_t align_up(size_t x, size_t a) { return (x + a - 1) & ~(a - 1); }

extern "C" void kernel_launch(void* const* d_in, const int* in_sizes, int n_in,
                              void* d_out, int out_size, void* d_ws, size_t ws_size,
                              hipStream_t stream) {
    (void)in_sizes; (void)n_in; (void)out_size; (void)ws_size;
    const float* x     = (const float*)d_in[0];
    const float* W_ih0 = (const float*)d_in[1];
    const float* b_ih0 = (const float*)d_in[2];
    const float* W_hh0 = (const float*)d_in[3];
    const float* b_hh0 = (const float*)d_in[4];
    const float* W_ih1 = (const float*)d_in[5];
    const float* b_ih1 = (const float*)d_in[6];
    const float* W_hh1 = (const float*)d_in[7];
    const float* b_hh1 = (const float*)d_in[8];
    float* out = (float*)d_out;

    // ---- carve workspace ----
    char* p = (char*)d_ws;
    size_t off = 0;
    auto carve = [&](size_t bytes) { void* r = p + off; off = align_up(off + bytes, 256); return r; };
    __bf16* xb    = (__bf16*)carve((size_t)Bq * Tq * INq * 2);
    __bf16* wih0b = (__bf16*)carve((size_t)GDIMq * INq * 2);
    __bf16* whh0b = (__bf16*)carve((size_t)GDIMq * Hq * 2);
    __bf16* wih1b = (__bf16*)carve((size_t)GDIMq * Hq * 2);
    __bf16* whh1b = (__bf16*)carve((size_t)GDIMq * Hq * 2);
    float*  tin   = (float*)carve((size_t)MROWS * GDIMq * 4);
    float*  gpart = (float*)carve((size_t)KSPLIT * Bq * GDIMq * 4);
    __bf16* ybf0  = (__bf16*)carve((size_t)Bq * (Tq + 1) * Hq * 2);
    __bf16* ybf1  = (__bf16*)carve((size_t)Bq * (Tq + 1) * Hq * 2);
    float*  c0    = (float*)carve((size_t)Bq * Hq * 4);
    float*  c1    = (float*)carve((size_t)Bq * Hq * 4);
    float*  hf0   = (float*)carve((size_t)Bq * Hq * 4);
    float*  hf1   = (float*)carve((size_t)Bq * Hq * 4);

    // ---- init ----
    zero_init_kernel<<<(Bq * Hq + 255) / 256, 256, 0, stream>>>(c0, c1, ybf0, ybf1);

    // ---- f32 -> bf16 conversions ----
    auto conv = [&](const float* s, __bf16* d, size_t n) {
        f32_to_bf16_kernel<<<(unsigned)((n + 255) / 256), 256, 0, stream>>>(s, d, (int)n);
    };
    conv(x,     xb,    (size_t)Bq * Tq * INq);
    conv(W_ih0, wih0b, (size_t)GDIMq * INq);
    conv(W_hh0, whh0b, (size_t)GDIMq * Hq);
    conv(W_ih1, wih1b, (size_t)GDIMq * Hq);
    conv(W_hh1, whh1b, (size_t)GDIMq * Hq);

    const int tinBlocks = MBLKS * NGRP2;                 // 256 * 17 = 4352
    const dim3 stepGrid((NTILES_G + 7) / 8, KSPLIT);     // (33, 4)

    // ================= layer 0 =================
    gemm_tin_kernel<<<tinBlocks, 256, 0, stream>>>(xb, wih0b, b_ih0, tin, INq, 0);
    for (int t = 0; t < Tq; ++t) {
        step_gemm_kernel<<<stepGrid, 256, 0, stream>>>(ybf0, whh0b, tin, gpart, t);
        gate_step_kernel<<<Bq, Hq, 0, stream>>>(gpart, tin, b_hh0, c0, hf0, ybf0, nullptr, t);
    }

    // ================= layer 1 =================
    // A-operand is layer-0's bf16 output sequence, stored with a leading zero column (skip=1)
    gemm_tin_kernel<<<tinBlocks, 256, 0, stream>>>(ybf0, wih1b, b_ih1, tin, Hq, 1);
    for (int t = 0; t < Tq; ++t) {
        step_gemm_kernel<<<stepGrid, 256, 0, stream>>>(ybf1, whh1b, tin, gpart, t);
        gate_step_kernel<<<Bq, Hq, 0, stream>>>(gpart, tin, b_hh1, c1, hf1, ybf1, out, t);
    }

    // ---- final states: [h0, h1] then [c0, c1] after the (B,T,H) sequence ----
    copy_out_kernel<<<(4 * Bq * Hq + 255) / 256, 256, 0, stream>>>(
        hf0, hf1, c0, c1, out + (size_t)Bq * Tq * Hq);
}
